// GNN_Model_67654324846600
// MI455X (gfx1250) — matrix operand order repaired and checked by reading, not compiled, not run on hardware
//
#include <hip/hip_runtime.h>

// ---------------------------------------------------------------------------
// GAT (3x GATConv + FC) for MI455X / gfx1250, wave32, WMMA bf16x3 GEMMs.
// GEMMs use v_wmma_f32_16x16x32_bf16 with an exact hi/lo bf16 split of both
// operands (3 WMMAs per tile) for ~fp32 accuracy. Edge softmax/aggregation is
// L2-resident atomics (working set ~130MB < 192MB L2).
// ---------------------------------------------------------------------------

typedef __attribute__((ext_vector_type(16))) __bf16 v16bf;
typedef __attribute__((ext_vector_type(8)))  float  v8f;

#ifndef NEG_SLOPE_C
#define NEG_SLOPE_C 0.2f
#endif

// ---------------------------------------------------------------------------
// WMMA GEMM:  Out[nrows, NCOLS] = X[nrows, KDIM] @ W[KDIM, NCOLS]
// One wave per 16-row tile, covering all NCOLS columns.
// bf16x3: acc += Ahi*Bhi + Alo*Bhi + Ahi*Blo  (f32 accumulate)
// ---------------------------------------------------------------------------
template<int KDIM, int NCOLS>
__global__ __launch_bounds__(256)
void gemm_bf16x3_k(const float* __restrict__ X, const float* __restrict__ W,
                   float* __restrict__ Out, int nrows) {
    constexpr int KT = (KDIM + 31) / 32;
    constexpr int NT = NCOLS / 16;

    const int lane = threadIdx.x & 31;
    const int wave = threadIdx.x >> 5;
    const long long mtile = (long long)blockIdx.x * (blockDim.x >> 5) + wave;
    const long long mBase = mtile * 16;
    if (mBase >= nrows) return;               // whole-wave exit: EXEC stays all-1 for WMMA

    const int ml = lane & 15;                 // row-in-tile (A) / col-in-tile (B,D)
    const int g  = lane >> 4;                 // lane group

    long long arow = mBase + ml;
    if (arow >= nrows) arow = nrows - 1;      // clamp (N % 16 == 0 in practice)
    const float* __restrict__ xrow = X + arow * (long long)KDIM;

    v8f acc[NT];
#pragma unroll
    for (int t = 0; t < NT; ++t) acc[t] = {};

#pragma unroll
    for (int kt = 0; kt < KT; ++kt) {
        const int k0 = kt * 32;
        // ---- A fragment (16x32 bf16, ISA layout): lane holds row ml, group g
        v16bf ahi, alo;
#pragma unroll
        for (int e = 0; e < 16; ++e) {
            const int v = e >> 1;
            const int k = k0 + ((v >= 4) ? 16 : 0) + 8 * g + ((v & 3) * 2) + (e & 1);
            float xv = (k < KDIM) ? xrow[k] : 0.0f;
            __bf16 h = (__bf16)xv;
            ahi[e] = h;
            alo[e] = (__bf16)(xv - (float)h);
        }
#pragma unroll
        for (int nt = 0; nt < NT; ++nt) {
            // ---- B fragment (32x16 bf16): lane holds column ml, k = 8g + e (e<8), 16 + 8g + (e-8) (e>=8)
            v16bf bhi, blo;
#pragma unroll
            for (int e = 0; e < 16; ++e) {
                const int k = k0 + ((e >= 8) ? 16 : 0) + 8 * g + (e & 7);
                float wv = (k < KDIM) ? W[(long long)k * NCOLS + nt * 16 + ml] : 0.0f;
                __bf16 h = (__bf16)wv;
                bhi[e] = h;
                blo[e] = (__bf16)(wv - (float)h);
            }
            acc[nt] = __builtin_amdgcn_wmma_f32_16x16x32_bf16(
                false, ahi, false, bhi, (short)0, acc[nt], false, false);
            acc[nt] = __builtin_amdgcn_wmma_f32_16x16x32_bf16(
                false, alo, false, bhi, (short)0, acc[nt], false, false);
            acc[nt] = __builtin_amdgcn_wmma_f32_16x16x32_bf16(
                false, ahi, false, blo, (short)0, acc[nt], false, false);
        }
    }

    // ---- D store (16x16 f32 layout: VGPR r -> m = r + 8g, n = ml)
#pragma unroll
    for (int nt = 0; nt < NT; ++nt) {
#pragma unroll
        for (int r = 0; r < 8; ++r) {
            const long long mm = mBase + r + 8 * g;
            if (mm < nrows)
                Out[mm * (long long)NCOLS + nt * 16 + ml] = acc[nt][r];
        }
    }
}

// ---------------------------------------------------------------------------
// Helpers / elementwise kernels
// ---------------------------------------------------------------------------
__global__ void fill_f32_k(float* __restrict__ p, float v, long long n) {
    long long i = (long long)blockIdx.x * blockDim.x + threadIdx.x;
    if (i < n) p[i] = v;
}

// a_s[n,h] = dot(xW[n,h,:], att_src[h,:]);  a_d likewise
template<int H, int C>
__global__ void asad_k(const float* __restrict__ Hm,
                       const float* __restrict__ att_s, const float* __restrict__ att_d,
                       float* __restrict__ as, float* __restrict__ ad, int N) {
    int tid = blockIdx.x * blockDim.x + threadIdx.x;
    if (tid >= N * H) return;
    const int n = tid / H, h = tid % H;
    const float* row = Hm + (long long)n * (H * C) + h * C;
    float ss = 0.f, sd = 0.f;
#pragma unroll
    for (int c = 0; c < C; ++c) {
        float v = row[c];
        ss += v * att_s[h * C + c];
        sd += v * att_d[h * C + c];
    }
    as[tid] = ss;
    ad[tid] = sd;
}

__device__ __forceinline__ void atomicMaxF(float* addr, float val) {
    // native-atomic float max: int max for non-negative, unsigned min for negative.
    // Valid when slot initialized to -inf (0xFF800000).
    if (val >= 0.0f) atomicMax((int*)addr, __float_as_int(val));
    else             atomicMin((unsigned int*)addr, __float_as_uint(val));
}

template<int H>
__device__ __forceinline__ void edge_sd(int e, int E, const int* __restrict__ src,
                                        const int* __restrict__ dst, int& s, int& d) {
    if (e < E) { s = src[e]; d = dst[e]; }
    else       { s = e - E; d = e - E; }      // appended self-loops
}

template<int H>
__global__ void edge_max_k(const int* __restrict__ src, const int* __restrict__ dst,
                           const float* __restrict__ as, const float* __restrict__ ad,
                           float* __restrict__ m, int E, int Etot) {
    long long tid = (long long)blockIdx.x * blockDim.x + threadIdx.x;
    if (tid >= (long long)Etot * H) return;
    const int e = (int)(tid / H), h = (int)(tid % H);
    int s, d; edge_sd<H>(e, E, src, dst, s, d);
    float v = as[s * H + h] + ad[d * H + h];
    v = (v > 0.f) ? v : NEG_SLOPE_C * v;
    atomicMaxF(&m[d * H + h], v);
}

template<int H>
__global__ void edge_sum_k(const int* __restrict__ src, const int* __restrict__ dst,
                           const float* __restrict__ as, const float* __restrict__ ad,
                           const float* __restrict__ m, float* __restrict__ pbuf,
                           float* __restrict__ denom, int E, int Etot) {
    long long tid = (long long)blockIdx.x * blockDim.x + threadIdx.x;
    if (tid >= (long long)Etot * H) return;
    const int e = (int)(tid / H), h = (int)(tid % H);
    int s, d; edge_sd<H>(e, E, src, dst, s, d);
    float v = as[s * H + h] + ad[d * H + h];
    v = (v > 0.f) ? v : NEG_SLOPE_C * v;
    float p = __expf(v - m[d * H + h]);
    pbuf[(long long)e * H + h] = p;
    atomicAdd(&denom[d * H + h], p);
}

template<int H, int NC>
__global__ void edge_agg_k(const int* __restrict__ src, const int* __restrict__ dst,
                           const float* __restrict__ Hm, const float* __restrict__ pbuf,
                           const float* __restrict__ denom, float* __restrict__ out,
                           int E, int Etot) {
    constexpr int C = NC / H;
    long long tid = (long long)blockIdx.x * blockDim.x + threadIdx.x;
    if (tid >= (long long)Etot * NC) return;
    const int e = (int)(tid / NC), c = (int)(tid % NC);
    const int h = c / C;
    int s, d; edge_sd<H>(e, E, src, dst, s, d);
    const float alpha = pbuf[(long long)e * H + h] / (denom[d * H + h] + 1e-16f);
    atomicAdd(&out[(long long)d * NC + c], Hm[(long long)s * NC + c] * alpha);
}

template<int NC>
__global__ void bias_relu_k(float* __restrict__ p, const float* __restrict__ b, long long n) {
    long long i = (long long)blockIdx.x * blockDim.x + threadIdx.x;
    if (i >= n) return;
    const int c = (int)(i % NC);
    float v = p[i] + b[c];
    p[i] = (v > 0.f) ? v : 0.f;
}

__global__ void fc_k(const float* __restrict__ Hm, const float* __restrict__ W,
                     const float* __restrict__ b, float* __restrict__ out, int N) {
    int n = blockIdx.x * blockDim.x + threadIdx.x;
    if (n >= N) return;
    float o0 = b[0], o1 = b[1];
    const float* row = Hm + (long long)n * 32;
#pragma unroll
    for (int c = 0; c < 32; ++c) {
        float v = row[c];
        o0 += v * W[c * 2 + 0];
        o1 += v * W[c * 2 + 1];
    }
    out[n * 2 + 0] = o0;
    out[n * 2 + 1] = o1;
}

// ---------------------------------------------------------------------------
// Host-side layer driver
// ---------------------------------------------------------------------------
static inline unsigned cdiv(long long a, long long b) { return (unsigned)((a + b - 1) / b); }

template<int KDIM, int NCOLS, int HEADSN>
static void gat_layer(const float* xin, const float* W, const float* att_s, const float* att_d,
                      const float* bias, const int* src, const int* dst,
                      int N, int E, int Etot,
                      float* xW, float* aggOut,
                      float* asb, float* adb, float* mb, float* db, float* pb,
                      hipStream_t stream) {
    constexpr int C = NCOLS / HEADSN;
    const int MT = (N + 15) / 16;

    gemm_bf16x3_k<KDIM, NCOLS><<<cdiv(MT, 8), 256, 0, stream>>>(xin, W, xW, N);
    asad_k<HEADSN, C><<<cdiv((long long)N * HEADSN, 256), 256, 0, stream>>>(
        xW, att_s, att_d, asb, adb, N);

    fill_f32_k<<<cdiv((long long)N * HEADSN, 256), 256, 0, stream>>>(
        mb, -__builtin_huge_valf(), (long long)N * HEADSN);
    fill_f32_k<<<cdiv((long long)N * HEADSN, 256), 256, 0, stream>>>(
        db, 0.0f, (long long)N * HEADSN);
    fill_f32_k<<<cdiv((long long)N * NCOLS, 256), 256, 0, stream>>>(
        aggOut, 0.0f, (long long)N * NCOLS);

    edge_max_k<HEADSN><<<cdiv((long long)Etot * HEADSN, 256), 256, 0, stream>>>(
        src, dst, asb, adb, mb, E, Etot);
    edge_sum_k<HEADSN><<<cdiv((long long)Etot * HEADSN, 256), 256, 0, stream>>>(
        src, dst, asb, adb, mb, pb, db, E, Etot);
    edge_agg_k<HEADSN, NCOLS><<<cdiv((long long)Etot * NCOLS, 256), 256, 0, stream>>>(
        src, dst, xW, pb, db, aggOut, E, Etot);

    bias_relu_k<NCOLS><<<cdiv((long long)N * NCOLS, 256), 256, 0, stream>>>(
        aggOut, bias, (long long)N * NCOLS);
}

extern "C" void kernel_launch(void* const* d_in, const int* in_sizes, int n_in,
                              void* d_out, int out_size, void* d_ws, size_t ws_size,
                              hipStream_t stream) {
    constexpr int IN_DIM = 122, HID = 32, HEADSN = 4;
    const int N    = in_sizes[0] / IN_DIM;
    const int E    = in_sizes[1] / 2;
    const int Etot = E + N;                     // with appended self-loops

    const float* x   = (const float*)d_in[0];
    const int*   ei  = (const int*)d_in[1];
    const float* W1  = (const float*)d_in[2];
    const float* a1s = (const float*)d_in[3];
    const float* a1d = (const float*)d_in[4];
    const float* b1  = (const float*)d_in[5];
    const float* W2  = (const float*)d_in[6];
    const float* a2s = (const float*)d_in[7];
    const float* a2d = (const float*)d_in[8];
    const float* b2  = (const float*)d_in[9];
    const float* W3  = (const float*)d_in[10];
    const float* a3s = (const float*)d_in[11];
    const float* a3d = (const float*)d_in[12];
    const float* b3  = (const float*)d_in[13];
    const float* Wfc = (const float*)d_in[14];
    const float* bfc = (const float*)d_in[15];

    const int* src = ei;
    const int* dst = ei + E;

    // Workspace layout (floats): ~136 MB total
    float* buf0 = (float*)d_ws;                       // [N,128] xW
    float* buf1 = buf0 + (long long)N * 128;          // [N,128] agg / next input
    float* asb  = buf1 + (long long)N * 128;          // [N,4]
    float* adb  = asb  + (long long)N * HEADSN;       // [N,4]
    float* mb   = adb  + (long long)N * HEADSN;       // [N,4]
    float* db   = mb   + (long long)N * HEADSN;       // [N,4]
    float* pb   = db   + (long long)N * HEADSN;       // [Etot,4]

    // conv1: 122 -> 4x32 (concat 128), relu     input = x, output -> buf1
    gat_layer<IN_DIM, HEADSN * HID, HEADSN>(x, W1, a1s, a1d, b1, src, dst, N, E, Etot,
                                            buf0, buf1, asb, adb, mb, db, pb, stream);
    // conv2: 128 -> 4x32 (concat 128), relu     input = buf1, xW -> buf0, output -> buf1
    gat_layer<HEADSN * HID, HEADSN * HID, HEADSN>(buf1, W2, a2s, a2d, b2, src, dst, N, E, Etot,
                                                  buf0, buf1, asb, adb, mb, db, pb, stream);
    // conv3: 128 -> 32 (1 head), relu           input = buf1, xW -> buf0, output -> buf1
    gat_layer<HEADSN * HID, HID, 1>(buf1, W3, a3s, a3d, b3, src, dst, N, E, Etot,
                                    buf0, buf1, asb, adb, mb, db, pb, stream);
    // fc: 32 -> 2
    fc_k<<<cdiv(N, 256), 256, 0, stream>>>(buf1, Wfc, bfc, (float*)d_out, N);
}
